// MLPDecoder_17643725652589
// MI455X (gfx1250) — matrix operand order repaired and checked by reading, not compiled, not run on hardware
//
#include <hip/hip_runtime.h>
#include <hip/hip_bf16.h>
#include <math.h>

// MI455X / gfx1250 implementation of the NRI-style MLP decoder.
// - One block per (batch, receiver): gather -> edge-MLP -> gate -> column-sum
//   fully fused; all GEMMs on v_wmma_f32_16x16x32_f16 (f16 operands, f32 acc).
// - Weights are pre-transposed/converted to K-major f16 ONCE (prep kernel),
//   then DMA'd into LDS per block with the Tensor Data Mover
//   (tensor_load_to_lds + s_wait_tensorcnt). This toolchain exposes the
//   6-arg clang-23 builtin form.

typedef __attribute__((ext_vector_type(16))) _Float16 v16h;
typedef __attribute__((ext_vector_type(8)))  float    v8f;
typedef __attribute__((ext_vector_type(4)))  unsigned int u32x4;
typedef __attribute__((ext_vector_type(8)))  int      i32x8;
typedef __attribute__((ext_vector_type(4)))  int      i32x4;

#define NN   256      // nodes
#define EE   65280    // edges = NN*(NN-1)
#define FF   16       // feature dim
#define MH   128      // hidden dims (MH == MO == NH == 128)
#define BB   4        // batch
#define PSTEPS 4      // pred_steps (reference fixes 4; T-1 == 4)

// K-major f16 weight blob layout in workspace (halves):
//   w1t [32][128]   @ 0          (4096)
//   w2t [128][128]  @ 4096       (16384)
//   w3t [160][128]  @ 20480      (20480, K rows 144..159 zero)
//   w4t [128][128]  @ 40960      (16384)
//   w5t [128][16]   @ 57344      (2048)
#define WB_W1T 0
#define WB_W2T 4096
#define WB_W3T 20480
#define WB_W4T 40960
#define WB_W5T 57344
#define WB_TOTAL 59392

#if defined(__gfx1250__) && __has_builtin(__builtin_amdgcn_tensor_load_to_lds) && \
    __has_builtin(__builtin_amdgcn_s_wait_tensorcnt)
#define USE_TDM 1
#else
#define USE_TDM 0
#endif

__device__ __forceinline__ v8f wmma16(v16h a, v16h b, v8f c) {
  // D = A(16x32 f16) x B(32x16 f16) + C(16x16 f32)
  return __builtin_amdgcn_wmma_f32_16x16x32_f16(false, a, false, b, (short)0, c,
                                                false, false);
}

// A-fragment (16x32 f16) from a row-major f16 LDS tile. ISA layout: lanes
// 0-15 hold M=lane, K {0..7,16..23}; lanes 16-31 hold M=lane-16,
// K {8..15,24..31}. Each 8-half run is contiguous -> ds_load_b128.
__device__ __forceinline__ v16h lds_a_frag(const _Float16* tile, int stride, int lane) {
  int m  = lane & 15;
  int kb = (lane & 16) ? 8 : 0;
  const _Float16* p = tile + m * stride + kb;
  v16h a;
#pragma unroll
  for (int i = 0; i < 8; ++i) a[i] = p[i];
#pragma unroll
  for (int i = 0; i < 8; ++i) a[8 + i] = p[16 + i];
  return a;
}

// B-fragment (32x16 f16) from K-major transposed weights wt[K][Ntot]:
// lane holds N=lane&15; lanes 0-15 K {0..7,16..23}, lanes 16-31 K {8..15,24..31}.
__device__ __forceinline__ v16h lds_b_frag(const _Float16* wt, int stride,
                                           int k0, int n0, int lane) {
  int kb = (lane & 16) ? 8 : 0;
  const _Float16* p = wt + (k0 + kb) * stride + n0 + (lane & 15);
  v16h b;
#pragma unroll
  for (int i = 0; i < 8; ++i) b[i] = p[i * stride];
#pragma unroll
  for (int i = 0; i < 8; ++i) b[8 + i] = p[(16 + i) * stride];
  return b;
}

#if USE_TDM
// 1D global->LDS DMA through the Tensor Data Mover. ndwords <= 65535.
// D# per CDNA5 ISA ch.8: group0 = {count=1, lds_addr, global_addr, type=2},
// group1 = {data_size=4B, tensor_dim0=tile_dim0=ndwords, dims1/2 unused}.
// clang-23 builtin: (u32x4 g0, i32x8 g1, i32x4 g2, i32x4 g3, i32x8, i32 cpol)
__device__ __forceinline__ void tdm_load_1d(const _Float16* gsrc,
                                            unsigned lds_byte_off,
                                            unsigned ndwords) {
  unsigned long long ga = (unsigned long long)(uintptr_t)gsrc;
  u32x4 g0;
  g0[0] = 1u;                                           // count=1, user D#
  g0[1] = lds_byte_off;                                 // lds_addr
  g0[2] = (unsigned)(ga & 0xffffffffu);                 // global_addr[31:0]
  g0[3] = (unsigned)((ga >> 32) & 0x01ffffffu) | (2u << 30);  // [56:32]|type=2
  i32x8 g1;
  g1[0] = (int)(2u << 16);                 // wg_mask=0, data_size=2 (4 bytes)
  g1[1] = (int)((ndwords & 0xffffu) << 16);       // tensor_dim0[15:0]
  g1[2] = (int)(((ndwords >> 16) & 0xffffu) | (1u << 16)); // td0 hi | td1=1
  g1[3] = (int)(ndwords << 16);                    // tile_dim0 = ndwords
  g1[4] = 1;                                       // tile_dim1=1, tile_dim2=0
  g1[5] = (int)ndwords;                            // tensor_dim0_stride lo
  g1[6] = 0;
  g1[7] = 0;
  i32x4 gz4 = {0, 0, 0, 0};
  i32x8 gz8 = {0, 0, 0, 0, 0, 0, 0, 0};
  __builtin_amdgcn_tensor_load_to_lds(g0, g1, gz4, gz4, gz8, 0);
}
#endif

// ---------------------------------------------------------------------------
// One-shot weight prep: f32 row-major (out,in) -> f16 K-major [K][N] blob.
__global__ void nri_prep_weights_kernel(const float* __restrict__ W1,
                                        const float* __restrict__ W2,
                                        const float* __restrict__ W3,
                                        const float* __restrict__ W4,
                                        const float* __restrict__ W5,
                                        _Float16* __restrict__ wb) {
  int idx = blockIdx.x * 256 + threadIdx.x;
  if (idx >= WB_TOTAL) return;
  _Float16 v;
  if (idx < WB_W2T) {                       // w1t [32][128] <- W1 (128,32)
    int q = idx - WB_W1T, k = q >> 7, n = q & 127;
    v = (_Float16)W1[n * 32 + k];
  } else if (idx < WB_W3T) {                // w2t [128][128] <- W2 (128,128)
    int q = idx - WB_W2T, k = q >> 7, n = q & 127;
    v = (_Float16)W2[n * 128 + k];
  } else if (idx < WB_W4T) {                // w3t [160][128] <- W3 (128,144)
    int q = idx - WB_W3T, k = q >> 7, n = q & 127;
    v = (k < 144) ? (_Float16)W3[n * 144 + k] : (_Float16)0.f;
  } else if (idx < WB_W5T) {                // w4t [128][128] <- W4 (128,128)
    int q = idx - WB_W4T, k = q >> 7, n = q & 127;
    v = (_Float16)W4[n * 128 + k];
  } else {                                  // w5t [128][16] <- W5 (16,128)
    int q = idx - WB_W5T, k = q >> 4, n = q & 15;
    v = (_Float16)W5[n * 128 + k];
  }
  wb[idx] = v;
}

__global__ void nri_init_last_kernel(const float* __restrict__ inputs,
                                     float* __restrict__ last) {
  int idx = blockIdx.x * 256 + threadIdx.x;   // 0 .. B*N*F-1 (16384)
  int f  = idx & 15;
  int bn = idx >> 4;                          // b*N + n
  // inputs (B,N,T=5,F): last[b][n][f] = inputs[b][n][0][f]
  last[idx] = inputs[bn * (5 * FF) + f];
}

// ---------------------------------------------------------------------------
// Edge MLP + aggregation. One block per (receiver j, batch b): 256 edge rows
// (sender i = row), row i==j masked via gate=0. 8 waves; wave w owns
// row-tiles 2w, 2w+1 through both GEMMs and the column reduction.
__global__ __launch_bounds__(256)
void nri_edge_mlp_agg_kernel(const float* __restrict__ last,      // (B,N,F)
                             const float* __restrict__ rel_type,  // (B,E,2)
                             const _Float16* __restrict__ wb,     // f16 blob
                             const float* __restrict__ b1,
                             const float* __restrict__ b2,
                             float* __restrict__ agg)             // (B,N,128)
{
  extern __shared__ char smem[];
  _Float16* lastf = (_Float16*)smem;            // 256*16
  _Float16* w1t   = lastf + NN * FF;            // 32*128   } contiguous,
  _Float16* w2t   = w1t + 32 * MH;              // 128*128  } one TDM tile
  _Float16* msgst = w2t + MH * MH;              // 8 waves * 16*128 staging
  float* gate_s = (float*)(msgst + 8 * 16 * MH);
  float* b1_s   = gate_s + NN;
  float* b2_s   = b1_s + MH;
  float* agg_s  = b2_s + MH;                    // 128

  const int j    = blockIdx.x;     // receiver node
  const int b    = blockIdx.y;     // batch
  const int t    = threadIdx.x;    // 0..255
  const int lane = t & 31;
  const int w    = t >> 5;         // wave id 0..7

  // ---- stage: weights via TDM (wave 0), features/gate by everyone ----
#if USE_TDM
  if (w == 0) {
    // w1t + w2t = 20480 halves = 10240 dwords, contiguous in blob and LDS
    tdm_load_1d(wb + WB_W1T, (unsigned)(uintptr_t)w1t, 10240u);
    __builtin_amdgcn_s_wait_tensorcnt(0);
  }
#else
  {
    const unsigned* src = (const unsigned*)(wb + WB_W1T);
    unsigned* dst = (unsigned*)w1t;
    for (int q = t; q < 10240; q += 256) dst[q] = src[q];
  }
#endif
  {
    const float* lg = last + (size_t)(b * NN) * FF;
#pragma unroll
    for (int f = 0; f < FF; ++f)
      lastf[t * FF + f] = (_Float16)lg[t * FF + f];
  }
  if (t < MH) { b1_s[t] = b1[t]; b2_s[t] = b2[t]; agg_s[t] = 0.f; }
  {
    int i = t;                                 // sender = row index
    float g = 0.f;
    if (i != j) {
      int e = i * (NN - 1) + (j > i ? j - 1 : j);
      g = rel_type[((size_t)b * EE + e) * 2 + 1];
    }
    gate_s[i] = g;
  }
  __syncthreads();

  const int n16 = lane & 15;
  const int hi  = (lane & 16) ? 8 : 0;

  // ---------------- GEMM1 (K=32, one WMMA per tile) ----------------
  v16h a2[2][4];
  _Float16* stg = msgst + w * (16 * MH);
#pragma unroll
  for (int rtl = 0; rtl < 2; ++rtl) {
    const int row0 = (w * 2 + rtl) * 16;
    // gather A fragment: pre_msg[row] = [lastf[row](send), lastf[j](recv)]
    v16h a1;
    {
      const _Float16* ps = lastf + (row0 + n16) * FF + hi;
      const _Float16* pr = lastf + j * FF + hi;
#pragma unroll
      for (int i = 0; i < 8; ++i) a1[i] = ps[i];
#pragma unroll
      for (int i = 0; i < 8; ++i) a1[8 + i] = pr[i];
    }
#pragma unroll
    for (int nt = 0; nt < 8; ++nt) {
      v16h bf = lds_b_frag(w1t, MH, 0, nt * 16, lane);
      float bias = b1_s[nt * 16 + n16];
      v8f acc;
#pragma unroll
      for (int r = 0; r < 8; ++r) acc[r] = bias;
      acc = wmma16(a1, bf, acc);
#pragma unroll
      for (int r = 0; r < 8; ++r)        // relu -> staging (A layout source)
        stg[(hi + r) * MH + nt * 16 + n16] = (_Float16)fmaxf(acc[r], 0.f);
    }
    asm volatile("s_wait_dscnt 0" ::: "memory");   // per-wave LDS RAW
#pragma unroll
    for (int kt = 0; kt < 4; ++kt)
      a2[rtl][kt] = lds_a_frag(stg + kt * 32, MH, lane);
  }

  // ------- GEMM2 (K=128, 4 chained WMMAs) + relu*gate + column sum -------
#pragma unroll 2
  for (int nt = 0; nt < 8; ++nt) {
    v16h bf[4];
#pragma unroll
    for (int kt = 0; kt < 4; ++kt)
      bf[kt] = lds_b_frag(w2t, MH, kt * 32, nt * 16, lane);
    float csum = 0.f;
    float bias = b2_s[nt * 16 + n16];
#pragma unroll
    for (int rtl = 0; rtl < 2; ++rtl) {
      v8f acc;
#pragma unroll
      for (int r = 0; r < 8; ++r) acc[r] = bias;
#pragma unroll
      for (int kt = 0; kt < 4; ++kt)
        acc = wmma16(a2[rtl][kt], bf[kt], acc);
      const int mbase = (w * 2 + rtl) * 16 + hi;
#pragma unroll
      for (int r = 0; r < 8; ++r)
        csum += fmaxf(acc[r], 0.f) * gate_s[mbase + r];
    }
    csum += __shfl_xor(csum, 16, 32);            // fold M halves (lane^16)
    if (lane < 16) atomicAdd(&agg_s[nt * 16 + n16], csum);  // ds_add_f32
  }
  __syncthreads();
  if (t < MH) agg[((size_t)b * NN + j) * MH + t] = agg_s[t];
}

// ---------------------------------------------------------------------------
// Node MLP: 128 node rows per block, 8 waves, wave w owns row-tile w.
// aug (144-wide) padded to K=160 for WMMA.
__global__ __launch_bounds__(256)
void nri_node_mlp_kernel(float* __restrict__ last,            // (B,N,16) rw
                         const float* __restrict__ agg,       // (B,N,128)
                         const _Float16* __restrict__ wb,     // f16 blob
                         const float* __restrict__ b3,
                         const float* __restrict__ b4,
                         const float* __restrict__ b5,
                         float* __restrict__ out,             // (B,N,4,16)
                         int step)
{
  extern __shared__ char smem[];
  _Float16* aug = (_Float16*)smem;          // 128*160
  _Float16* w3t = aug + 128 * 160;          // 160*128 } contiguous,
  _Float16* w4t = w3t + 160 * MH;           // 128*128 } one TDM tile
  _Float16* w5t = w4t + MH * MH;            // 128*16  }
  _Float16* h1  = w5t + MH * 16;            // 128*128
  _Float16* h2  = h1 + 128 * MH;            // 128*128
  float* b3_s = (float*)(h2 + 128 * MH);
  float* b4_s = b3_s + MH;
  float* b5_s = b4_s + MH;

  const int t    = threadIdx.x;
  const int lane = t & 31;
  const int w    = t >> 5;
  const int g0   = blockIdx.x * 128;        // first global node-row (b*N+n)

  // ---------------- stage ----------------
#if USE_TDM
  if (w == 0) {
    // w3t + w4t + w5t = 38912 halves = 19456 dwords
    tdm_load_1d(wb + WB_W3T, (unsigned)(uintptr_t)w3t, 19456u);
    __builtin_amdgcn_s_wait_tensorcnt(0);
  }
#else
  {
    const unsigned* src = (const unsigned*)(wb + WB_W3T);
    unsigned* dst = (unsigned*)w3t;
    for (int q = t; q < 19456; q += 256) dst[q] = src[q];
  }
#endif
  if (t < 128) {
    int g = g0 + t;
#pragma unroll
    for (int f = 0; f < FF; ++f)
      aug[t * 160 + f] = (_Float16)last[(size_t)g * FF + f];
#pragma unroll 8
    for (int c = 0; c < MH; ++c)
      aug[t * 160 + FF + c] = (_Float16)agg[(size_t)g * MH + c];
#pragma unroll
    for (int p = 0; p < 16; ++p)
      aug[t * 160 + 144 + p] = (_Float16)0.f;
  }
  if (t < MH) { b3_s[t] = b3[t]; b4_s[t] = b4[t]; }
  if (t < 16) b5_s[t] = b5[t];
  __syncthreads();

  const int n16  = lane & 15;
  const int hi   = (lane & 16) ? 8 : 0;
  const int row0 = w * 16;

  // h1 = relu(aug @ W3^T + b3), K = 160 (5 tiles)
#pragma unroll 2
  for (int nt = 0; nt < 8; ++nt) {
    float bias = b3_s[nt * 16 + n16];
    v8f acc;
#pragma unroll
    for (int r = 0; r < 8; ++r) acc[r] = bias;
#pragma unroll
    for (int kt = 0; kt < 5; ++kt) {
      v16h a  = lds_a_frag(aug + row0 * 160 + kt * 32, 160, lane);
      v16h bf = lds_b_frag(w3t, MH, kt * 32, nt * 16, lane);
      acc = wmma16(a, bf, acc);
    }
#pragma unroll
    for (int r = 0; r < 8; ++r)
      h1[(row0 + hi + r) * MH + nt * 16 + n16] = (_Float16)fmaxf(acc[r], 0.f);
  }
  asm volatile("s_wait_dscnt 0" ::: "memory");

  // h2 = relu(h1 @ W4^T + b4), K = 128
#pragma unroll 2
  for (int nt = 0; nt < 8; ++nt) {
    float bias = b4_s[nt * 16 + n16];
    v8f acc;
#pragma unroll
    for (int r = 0; r < 8; ++r) acc[r] = bias;
#pragma unroll
    for (int kt = 0; kt < 4; ++kt) {
      v16h a  = lds_a_frag(h1 + row0 * MH + kt * 32, MH, lane);
      v16h bf = lds_b_frag(w4t, MH, kt * 32, nt * 16, lane);
      acc = wmma16(a, bf, acc);
    }
#pragma unroll
    for (int r = 0; r < 8; ++r)
      h2[(row0 + hi + r) * MH + nt * 16 + n16] = (_Float16)fmaxf(acc[r], 0.f);
  }
  asm volatile("s_wait_dscnt 0" ::: "memory");

  // pred = tanh(h2 @ W5^T + b5): single 16-col tile, K = 128
  {
    float bias = b5_s[n16];
    v8f acc;
#pragma unroll
    for (int r = 0; r < 8; ++r) acc[r] = bias;
#pragma unroll
    for (int kt = 0; kt < 4; ++kt) {
      v16h a  = lds_a_frag(h2 + row0 * MH + kt * 32, MH, lane);
      v16h bf = lds_b_frag(w5t, 16, kt * 32, 0, lane);
      acc = wmma16(a, bf, acc);
    }
#pragma unroll
    for (int r = 0; r < 8; ++r) {
      int g = g0 + row0 + hi + r;               // global node row = b*N+n
      float nv = last[(size_t)g * FF + n16] + tanhf(acc[r]);
      last[(size_t)g * FF + n16] = nv;
      out[((size_t)g * PSTEPS + step) * FF + n16] = nv;   // (B,N,4,16)
    }
  }
}

extern "C" void kernel_launch(void* const* d_in, const int* in_sizes, int n_in,
                              void* d_out, int out_size, void* d_ws, size_t ws_size,
                              hipStream_t stream) {
  const float* inputs   = (const float*)d_in[0];   // (4,256,5,16)
  const float* rel_type = (const float*)d_in[1];   // (4,65280,2)
  // d_in[2] rel_rec, d_in[3] rel_send: all-ordered-pairs structure exploited
  // analytically (e = i*255 + (j>i ? j-1 : j)); not read.
  const float* W1 = (const float*)d_in[4];
  const float* b1 = (const float*)d_in[5];
  const float* W2 = (const float*)d_in[6];
  const float* b2 = (const float*)d_in[7];
  const float* W3 = (const float*)d_in[8];
  const float* b3 = (const float*)d_in[9];
  const float* W4 = (const float*)d_in[10];
  const float* b4 = (const float*)d_in[11];
  const float* W5 = (const float*)d_in[12];
  const float* b5 = (const float*)d_in[13];
  // d_in[14] = pred_steps scalar; reference fixes 4 (out_size == 65536).

  float* out  = (float*)d_out;
  float* last = (float*)d_ws;                          // B*N*F   f32
  float* agg  = (float*)((char*)d_ws + 16384 * 4);     // B*N*128 f32
  _Float16* wb = (_Float16*)((char*)d_ws + 16384 * 4 + 131072 * 4);

  nri_prep_weights_kernel<<<(WB_TOTAL + 255) / 256, 256, 0, stream>>>(
      W1, W2, W3, W4, W5, wb);
  nri_init_last_kernel<<<64, 256, 0, stream>>>(inputs, last);

  const size_t edge_lds =
      (size_t)(NN * FF + 32 * MH + MH * MH + 8 * 16 * MH) * sizeof(_Float16) +
      (size_t)(NN + MH + MH + MH) * sizeof(float);               // ~84.5 KB
  const size_t node_lds =
      (size_t)(128 * 160 + 160 * MH + MH * MH + MH * 16 + 2 * 128 * MH) *
          sizeof(_Float16) +
      (size_t)(MH + MH + 16) * sizeof(float);                    // ~182 KB

  for (int s = 0; s < PSTEPS; ++s) {
    dim3 eg(NN, BB);   // 1024 blocks: one per (receiver, batch)
    nri_edge_mlp_agg_kernel<<<eg, 256, edge_lds, stream>>>(
        last, rel_type, wb, b1, b2, agg);
    nri_node_mlp_kernel<<<8, 256, node_lds, stream>>>(
        last, agg, wb, b3, b4, b5, out, s);
  }
}
